// GAT_Temporal_Regressor_31327491457264
// MI455X (gfx1250) — compile-verified, matrix operand order checked by reading, not run on hardware
//
#include <hip/hip_runtime.h>
#include <hip/hip_bf16.h>

typedef __attribute__((ext_vector_type(16))) _Float16 v16h;
typedef __attribute__((ext_vector_type(8)))  float    v8f;

#define NEG_SLOPE 0.2f

// ---------------- problem constants (from reference setup_inputs) -------------
#define NB   2        // batch
#define NW   12       // time steps in input
#define NN   10000    // nodes per batch
#define NF   16       // input features
#define NHID 32       // hidden per head
#define NH   2        // heads
#define NM   (NB*NN)          // 20000 merged nodes
#define NT   2                // only t=10,11 feed the output
#define T0   10               // first live time step
#define NR   (NT*NM)          // 40000 GEMM rows

// ---------------------------------------------------------------------------
// float atomic max via signed/unsigned integer atomics
__device__ __forceinline__ void atomicMaxFloat(float* addr, float v) {
    if (v >= 0.0f) atomicMax((int*)addr, __float_as_int(v));
    else           atomicMin((unsigned int*)addr, __float_as_uint(v));
}

__device__ __forceinline__ void edge_sd(const long long* ei, long long Eraw,
                                        long long e, int& s, int& d) {
    if (e < Eraw) { s = (int)ei[e]; d = (int)ei[Eraw + e]; }
    else          { s = d = (int)(e - Eraw); }          // appended self loops
}

// ---------------- weight / input conversion to f16 --------------------------
// W1t: [64][32] col-major (n-major), K padded 16->32 with zeros
// W2t: [64][64] col-major
__global__ void k_convert_w(const float* __restrict__ W1,
                            const float* __restrict__ W2,
                            _Float16* __restrict__ W1t,
                            _Float16* __restrict__ W2t) {
    int n = threadIdx.x;           // 64 threads
    for (int k = 0; k < 32; ++k)
        W1t[n*32 + k] = (k < NF) ? (_Float16)W1[k*64 + n] : (_Float16)0.0f;
    for (int k = 0; k < 64; ++k)
        W2t[n*64 + k] = (_Float16)W2[k*64 + n];
}

// A1h: [NT*NM][32] f16 row-major, features padded 16->32 with zeros
__global__ void k_convert_x(const float* __restrict__ x, _Float16* __restrict__ A1h) {
    int tid = blockIdx.x * blockDim.x + threadIdx.x;
    if (tid >= NR) return;
    int tt = tid / NM, m = tid % NM;
    int b = m / NN, n = m % NN;
    const float* xr = x + (((size_t)(b*NW + (T0+tt))*NN + n) * NF);
    _Float16* a = A1h + (size_t)tid * 32;
    #pragma unroll
    for (int f = 0; f < NF; ++f) a[f] = (_Float16)xr[f];
    #pragma unroll
    for (int f = NF; f < 32; ++f) a[f] = (_Float16)0.0f;
}

// ---------------- WMMA GEMM: C[rows][64] = A[rows][KT*32] * Bt^T -------------
// Bt stored column-major: Bt[n][K]. Block = 128 thr = 4 waves, wave w -> n-tile w.
// 16-bit A 16x32 lane layout per CDNA5 ISA: lane = (row%16) + 16*half,
//   VGPR v<4 : K = 2v + 8*half ; v>=4 : K = 16 + 2(v-4) + 8*half  (pairs packed)
template<int KT>
__global__ void k_gemm_wmma(const _Float16* __restrict__ A,
                            const _Float16* __restrict__ Bt,
                            float* __restrict__ C, int rows) {
    const int wave = threadIdx.x >> 5;
    const int lane = threadIdx.x & 31;
    const int r    = lane & 15;
    const int half = lane >> 4;
    const int row0 = blockIdx.x * 16;
    const int K    = KT * 32;

    const _Float16* arow = A  + (size_t)(row0 + r) * K;
    const _Float16* bcol = Bt + (size_t)(wave*16 + r) * K;

    v8f acc = {};
    #pragma unroll
    for (int kt = 0; kt < KT; ++kt) {
        v16h av, bv;
        const int kb = kt * 32;
        #pragma unroll
        for (int v = 0; v < 8; ++v) {
            int k0 = kb + ((v < 4) ? (2*v + 8*half) : (16 + 2*(v-4) + 8*half));
            av[2*v]   = arow[k0];
            av[2*v+1] = arow[k0+1];
            bv[2*v]   = bcol[k0];
            bv[2*v+1] = bcol[k0+1];
        }
        acc = __builtin_amdgcn_wmma_f32_16x16x32_f16(
                  false, av, false, bv, (short)0, acc, false, false);
    }
    // D layout: VGPR v -> M = v + 8*half, N = lane%16
    float* cr = C + (size_t)row0 * 64 + wave*16 + r;
    #pragma unroll
    for (int v = 0; v < 8; ++v)
        cr[(size_t)(v + 8*half) * 64] = acc[v];
}

// ---------------- attention scalars a_s, a_d per (row, head) -----------------
__global__ void k_att_vec(const float* __restrict__ h,
                          const float* __restrict__ attS,
                          const float* __restrict__ attD,
                          float* __restrict__ asv, float* __restrict__ adv) {
    int tid = blockIdx.x * blockDim.x + threadIdx.x;
    if (tid >= NR * NH) return;
    int row = tid >> 1, hh = tid & 1;
    const float* hr = h + (size_t)row * 64 + hh * NHID;
    const float* s  = attS + hh * NHID;
    const float* d  = attD + hh * NHID;
    float as = 0.f, ad = 0.f;
    #pragma unroll
    for (int c = 0; c < NHID; ++c) { as += hr[c] * s[c]; ad += hr[c] * d[c]; }
    asv[tid] = as; adv[tid] = ad;
}

// ---------------- per-layer edge-softmax scratch init ------------------------
__global__ void k_init_edge(float* __restrict__ agg, float* __restrict__ mbuf,
                            float* __restrict__ sbuf) {
    int tid = blockIdx.x * blockDim.x + threadIdx.x;
    if (tid < NR * 64) agg[tid] = 0.0f;
    if (tid < NR * NH) { mbuf[tid] = -__builtin_huge_valf(); sbuf[tid] = 0.0f; }
}

// ---------------- segment max over edges -------------------------------------
__global__ void k_edge_max(const long long* __restrict__ ei, long long Eraw, long long Etot,
                           const float* __restrict__ asv, const float* __restrict__ adv,
                           float* __restrict__ mbuf) {
    long long tid = (long long)blockIdx.x * blockDim.x + threadIdx.x;
    if (tid >= Etot * NT) return;
    long long e = tid % Etot; int tt = (int)(tid / Etot);
    int s, d; edge_sd(ei, Eraw, e, s, d);
    int rs = tt * NM + s, rd = tt * NM + d;
    #pragma unroll
    for (int hh = 0; hh < NH; ++hh) {
        float ev = asv[rs*NH + hh] + adv[rd*NH + hh];
        ev = (ev > 0.0f) ? ev : NEG_SLOPE * ev;
        atomicMaxFloat(&mbuf[rd*NH + hh], ev);
    }
}

// ---------------- segment sum of exp -----------------------------------------
__global__ void k_edge_sum(const long long* __restrict__ ei, long long Eraw, long long Etot,
                           const float* __restrict__ asv, const float* __restrict__ adv,
                           const float* __restrict__ mbuf, float* __restrict__ sbuf) {
    long long tid = (long long)blockIdx.x * blockDim.x + threadIdx.x;
    if (tid >= Etot * NT) return;
    long long e = tid % Etot; int tt = (int)(tid / Etot);
    int s, d; edge_sd(ei, Eraw, e, s, d);
    int rs = tt * NM + s, rd = tt * NM + d;
    #pragma unroll
    for (int hh = 0; hh < NH; ++hh) {
        float ev = asv[rs*NH + hh] + adv[rd*NH + hh];
        ev = (ev > 0.0f) ? ev : NEG_SLOPE * ev;
        atomicAdd(&sbuf[rd*NH + hh], __expf(ev - mbuf[rd*NH + hh]));
    }
}

// ---------------- weighted message aggregation -------------------------------
// one 32-lane group per (tt, head, edge); lane = channel
__global__ void k_edge_agg(const long long* __restrict__ ei, long long Eraw, long long Etot,
                           const float* __restrict__ asv, const float* __restrict__ adv,
                           const float* __restrict__ mbuf, const float* __restrict__ sbuf,
                           const float* __restrict__ h, float* __restrict__ agg) {
    long long tid = (long long)blockIdx.x * blockDim.x + threadIdx.x;
    int lane = (int)(tid & 31);
    long long w = tid >> 5;
    if (w >= Etot * NT * NH) return;
    long long e = w % Etot;
    int hh = (int)((w / Etot) % NH);
    int tt = (int)(w / (Etot * NH));
    int s, d; edge_sd(ei, Eraw, e, s, d);
    int rs = tt * NM + s, rd = tt * NM + d;
    float ev = asv[rs*NH + hh] + adv[rd*NH + hh];
    ev = (ev > 0.0f) ? ev : NEG_SLOPE * ev;
    float alpha = __expf(ev - mbuf[rd*NH + hh]) / sbuf[rd*NH + hh];
    atomicAdd(&agg[(size_t)rd*64 + hh*NHID + lane],
              alpha * h[(size_t)rs*64 + hh*NHID + lane]);
}

// ---------------- layer1 post: bias + relu -> f16 A for GEMM2 ---------------
__global__ void k_post1(const float* __restrict__ agg, const float* __restrict__ b1,
                        _Float16* __restrict__ A2h) {
    int tid = blockIdx.x * blockDim.x + threadIdx.x;
    if (tid >= NR * 64) return;
    float v = agg[tid] + b1[tid & 63];
    v = fmaxf(v, 0.0f);
    A2h[tid] = (_Float16)v;
}

// ---------------- layer2 post: head mean + bias + relu ----------------------
__global__ void k_post2(const float* __restrict__ agg, const float* __restrict__ b2,
                        float* __restrict__ h2) {
    int tid = blockIdx.x * blockDim.x + threadIdx.x;
    if (tid >= NR * NHID) return;
    int row = tid >> 5, c = tid & 31;
    float v = 0.5f * (agg[(size_t)row*64 + c] + agg[(size_t)row*64 + 32 + c]) + b2[c];
    h2[tid] = fmaxf(v, 0.0f);
}

// ---------------- temporal conv (last column only) + FC ---------------------
// out[t=11] uses k=0 -> h2(t=10), k=1 -> h2(t=11); k=2 hits zero padding
__global__ void k_final(const float* __restrict__ h2, const float* __restrict__ wconv,
                        const float* __restrict__ bconv, const float* __restrict__ wfc,
                        const float* __restrict__ bfc, float* __restrict__ out) {
    int m = blockIdx.x * blockDim.x + threadIdx.x;
    if (m >= NM) return;
    const float* h0 = h2 + (size_t)(0*NM + m) * NHID;   // t = 10
    const float* h1 = h2 + (size_t)(1*NM + m) * NHID;   // t = 11
    float acc = bfc[0];
    #pragma unroll
    for (int o = 0; o < 32; ++o) {
        float to = bconv[o];
        const float* wr = wconv + (size_t)o * NHID * 3;
        #pragma unroll
        for (int i = 0; i < NHID; ++i)
            to += h0[i] * wr[i*3 + 0] + h1[i] * wr[i*3 + 1];
        acc += fmaxf(to, 0.0f) * wfc[o];
    }
    out[m] = acc;
}

// ===========================================================================
extern "C" void kernel_launch(void* const* d_in, const int* in_sizes, int n_in,
                              void* d_out, int out_size, void* d_ws, size_t ws_size,
                              hipStream_t stream) {
    const float*     x   = (const float*)d_in[0];
    const long long* ei  = (const long long*)d_in[1];
    const float*     W1  = (const float*)d_in[2];
    const float*     as1 = (const float*)d_in[3];
    const float*     ad1 = (const float*)d_in[4];
    const float*     b1  = (const float*)d_in[5];
    const float*     W2  = (const float*)d_in[6];
    const float*     as2 = (const float*)d_in[7];
    const float*     ad2 = (const float*)d_in[8];
    const float*     b2  = (const float*)d_in[9];
    const float*     wconv = (const float*)d_in[10];
    const float*     bconv = (const float*)d_in[11];
    const float*     wfc   = (const float*)d_in[12];
    const float*     bfc   = (const float*)d_in[13];
    float* out = (float*)d_out;

    const long long Eraw = in_sizes[1] / 2;      // 320000
    const long long Etot = Eraw + NM;            // + self loops

    // ---- workspace carve-out ----
    char* ws = (char*)d_ws;
    size_t off = 0;
    auto carve = [&](size_t bytes) {
        off = (off + 255) & ~(size_t)255;
        void* p = ws + off; off += bytes; return p;
    };
    _Float16* A1h  = (_Float16*)carve((size_t)NR * 32 * sizeof(_Float16));
    _Float16* A2h  = (_Float16*)carve((size_t)NR * 64 * sizeof(_Float16));
    _Float16* W1t  = (_Float16*)carve((size_t)64 * 32 * sizeof(_Float16));
    _Float16* W2t  = (_Float16*)carve((size_t)64 * 64 * sizeof(_Float16));
    float*    hbuf = (float*)carve((size_t)NR * 64 * sizeof(float));  // h1 then h2lin
    float*    asv  = (float*)carve((size_t)NR * NH * sizeof(float));
    float*    adv  = (float*)carve((size_t)NR * NH * sizeof(float));
    float*    mbuf = (float*)carve((size_t)NR * NH * sizeof(float));
    float*    sbuf = (float*)carve((size_t)NR * NH * sizeof(float));
    float*    agg  = (float*)carve((size_t)NR * 64 * sizeof(float));
    float*    h2   = (float*)carve((size_t)NR * NHID * sizeof(float));
    (void)ws_size; (void)n_in; (void)out_size;

    const int TB = 256;
    auto blks = [](long long n, int tb) { return (unsigned)((n + tb - 1) / tb); };
    const long long nET  = Etot * NT;
    const long long nAGG = Etot * NT * NH * 32;

    // ---- stage inputs as f16 ----
    k_convert_w<<<1, 64, 0, stream>>>(W1, W2, W1t, W2t);
    k_convert_x<<<blks(NR, TB), TB, 0, stream>>>(x, A1h);

    // ---- layer 1: WMMA GEMM (K=32, padded from 16) ----
    k_gemm_wmma<1><<<NR / 16, 128, 0, stream>>>(A1h, W1t, hbuf, NR);
    k_att_vec<<<blks((long long)NR * NH, TB), TB, 0, stream>>>(hbuf, as1, ad1, asv, adv);
    k_init_edge<<<blks((long long)NR * 64, TB), TB, 0, stream>>>(agg, mbuf, sbuf);
    k_edge_max<<<blks(nET, TB), TB, 0, stream>>>(ei, Eraw, Etot, asv, adv, mbuf);
    k_edge_sum<<<blks(nET, TB), TB, 0, stream>>>(ei, Eraw, Etot, asv, adv, mbuf, sbuf);
    k_edge_agg<<<blks(nAGG, TB), TB, 0, stream>>>(ei, Eraw, Etot, asv, adv, mbuf, sbuf, hbuf, agg);
    k_post1<<<blks((long long)NR * 64, TB), TB, 0, stream>>>(agg, b1, A2h);

    // ---- layer 2: WMMA GEMM (K=64 -> 2 wmma steps) ----
    k_gemm_wmma<2><<<NR / 16, 128, 0, stream>>>(A2h, W2t, hbuf, NR);
    k_att_vec<<<blks((long long)NR * NH, TB), TB, 0, stream>>>(hbuf, as2, ad2, asv, adv);
    k_init_edge<<<blks((long long)NR * 64, TB), TB, 0, stream>>>(agg, mbuf, sbuf);
    k_edge_max<<<blks(nET, TB), TB, 0, stream>>>(ei, Eraw, Etot, asv, adv, mbuf);
    k_edge_sum<<<blks(nET, TB), TB, 0, stream>>>(ei, Eraw, Etot, asv, adv, mbuf, sbuf);
    k_edge_agg<<<blks(nAGG, TB), TB, 0, stream>>>(ei, Eraw, Etot, asv, adv, mbuf, sbuf, hbuf, agg);
    k_post2<<<blks((long long)NR * NHID, TB), TB, 0, stream>>>(agg, b2, h2);

    // ---- temporal conv (last column) + FC head ----
    k_final<<<blks(NM, TB), TB, 0, stream>>>(h2, wconv, bconv, wfc, bfc, out);
}
